// Model_70007966925431
// MI455X (gfx1250) — matrix operand order repaired and checked by reading, not compiled, not run on hardware
//
#include <hip/hip_runtime.h>
#include <hip/hip_bf16.h>
#include <hip/hip_fp16.h>

typedef __attribute__((ext_vector_type(16))) _Float16 v16h;
typedef __attribute__((ext_vector_type(8)))  _Float16 v8h;
typedef __attribute__((ext_vector_type(8)))  float    v8f;

#define NQ    16      // query images
#define NN    64      // normal images
#define PSZ   16      // patch size
#define PPI   196     // patches per image (14x14)
#define PPAD  208     // padded patches per image (13*16)
#define DD    768     // feature dim (= 3*16*16 = EMBED)
#define RA    3136    // NQ*PPI
#define RBPAD 13312   // NN*PPAD
#define RALL  15680   // (NQ+NN)*PPI
#define HWH   224
#define PHD   14

// ---------------- WMMA fragment loaders (ISA 7.12.2 layouts) ----------------
// A 16x32 f16: lanes 0-15 row M=lane, K 0-7 (v0-3) + 16-23 (v4-7);
//              lanes 16-31 row M=lane-16, K 8-15 + 24-31.
__device__ __forceinline__ v16h load_a_frag(const _Float16* __restrict__ A,
                                            int ld, int row0, int k0) {
  int lane = threadIdx.x & 31;
  int r = lane & 15;
  int koff = (lane & 16) ? 8 : 0;
  const _Float16* p = A + (size_t)(row0 + r) * ld + (k0 + koff);
  v8h lo = *(const v8h*)p;          // K +0..7   (16B aligned)
  v8h hi = *(const v8h*)(p + 16);   // K +16..23
  v16h o;
#pragma unroll
  for (int i = 0; i < 8; ++i) { o[i] = lo[i]; o[i + 8] = hi[i]; }
  return o;
}
// B 32x16 f16 from row-major Bt[N][K] (Bt row n == B column n):
// lanes 0-15 col N=lane, K 0-7 + 16-23; lanes 16-31 col N=lane-16, K 8-15 + 24-31.
__device__ __forceinline__ v16h load_b_frag(const _Float16* __restrict__ Bt,
                                            int ld, int col0, int k0) {
  int lane = threadIdx.x & 31;
  int c = lane & 15;
  int koff = (lane & 16) ? 8 : 0;
  const _Float16* p = Bt + (size_t)(col0 + c) * ld + (k0 + koff);
  v8h lo = *(const v8h*)p;
  v8h hi = *(const v8h*)(p + 16);
  v16h o;
#pragma unroll
  for (int i = 0; i < 8; ++i) { o[i] = lo[i]; o[i + 8] = hi[i]; }
  return o;
}

// ---------------- prep kernels ----------------
__global__ void init_kernel(unsigned int* __restrict__ sp,
                            float* __restrict__ sqb,
                            _Float16* __restrict__ Bpad) {
  size_t i = (size_t)blockIdx.x * blockDim.x + threadIdx.x;
  size_t stride = (size_t)gridDim.x * blockDim.x;
  for (size_t k = i; k < (size_t)RA * NN; k += stride) sp[k] = 0x7F800000u; // +inf
  for (size_t k = i; k < (size_t)RBPAD; k += stride) sqb[k] = 1e30f;       // pad rows stay huge
  for (size_t k = i; k < (size_t)RBPAD * DD; k += stride) Bpad[k] = (_Float16)0.0f;
}

__global__ void patchify_kernel(const float* __restrict__ images,
                                const float* __restrict__ normals,
                                _Float16* __restrict__ Pall) {
  size_t idx = (size_t)blockIdx.x * blockDim.x + threadIdx.x;
  if (idx >= (size_t)RALL * DD) return;
  int row = (int)(idx / DD);
  int f   = (int)(idx % DD);
  int n = row / PPI, pp = row % PPI;
  int ph = pp / PHD, pw = pp % PHD;
  int c = f >> 8, r2 = f & 255, py = r2 >> 4, px = r2 & 15;
  const float* src = (n < NQ)
      ? images  + (((size_t)n        * 3 + c) * HWH * HWH)
      : normals + (((size_t)(n - NQ) * 3 + c) * HWH * HWH);
  Pall[idx] = (_Float16)src[(ph * PSZ + py) * HWH + (pw * PSZ + px)];
}

__global__ void wtrans_kernel(const float* __restrict__ W, _Float16* __restrict__ Wt) {
  int idx = blockIdx.x * blockDim.x + threadIdx.x;
  if (idx >= DD * DD) return;
  int n = idx / DD, k = idx % DD;
  Wt[(size_t)n * DD + k] = (_Float16)W[(size_t)k * DD + n];
}

// ---------------- GEMM 1: F = Pall @ W  (f16 x f16 -> f32) ----------------
// one wave (32 threads) per 16x64 output strip
__global__ void proj_gemm_kernel(const _Float16* __restrict__ A,
                                 const _Float16* __restrict__ Bt,
                                 float* __restrict__ C) {
  int row0 = blockIdx.x * 16;
  int col0 = blockIdx.y * 64;
  v8f acc[4] = {};
  for (int k0 = 0; k0 < DD; k0 += 32) {
    v16h a = load_a_frag(A, DD, row0, k0);
#pragma unroll
    for (int j = 0; j < 4; ++j) {
      v16h b = load_b_frag(Bt, DD, col0 + j * 16, k0);
      acc[j] = __builtin_amdgcn_wmma_f32_16x16x32_f16(false, a, false, b,
                                                      (short)0, acc[j], false, false);
    }
  }
  int lane = threadIdx.x & 31;
  int c = lane & 15;
  int rbase = (lane & 16) ? 8 : 0;
#pragma unroll
  for (int j = 0; j < 4; ++j)
#pragma unroll
    for (int r = 0; r < 8; ++r)
      C[(size_t)(row0 + rbase + r) * DD + (col0 + j * 16 + c)] = acc[j][r];
}

// ---------------- row-normalize -> f16 A / padded B ----------------
__global__ void normalize_kernel(const float* __restrict__ F,
                                 _Float16* __restrict__ A,
                                 _Float16* __restrict__ Bpad,
                                 float* __restrict__ sqa,
                                 float* __restrict__ sqb) {
  int row = blockIdx.x;        // 0..RALL-1
  int t = threadIdx.x;         // 256
  __shared__ float red[256];
  const float* src = F + (size_t)row * DD;
  float s = 0.f;
  for (int i = t; i < DD; i += 256) { float v = src[i]; s += v * v; }
  red[t] = s; __syncthreads();
  for (int off = 128; off > 0; off >>= 1) {
    if (t < off) red[t] += red[t + off];
    __syncthreads();
  }
  float rn = rsqrtf(fmaxf(red[0], 1e-24f));
  _Float16* dst;
  if (row < RA) {
    dst = A + (size_t)row * DD;
    if (t == 0) sqa[row] = 1.0f;                 // normalized -> ||a||^2 = 1
  } else {
    int rr = row - RA;
    int prow = (rr / PPI) * PPAD + (rr % PPI);   // place into padded layout
    dst = Bpad + (size_t)prow * DD;
    if (t == 0) sqb[prow] = 1.0f;
  }
  for (int i = t; i < DD; i += 256) dst[i] = (_Float16)(src[i] * rn);
}

// ---------------- GEMM 2: dist + fused per-image min ----------------
__global__ void dist_gemm_kernel(const _Float16* __restrict__ A,
                                 const _Float16* __restrict__ Bt,
                                 const float* __restrict__ sqa,
                                 const float* __restrict__ sqb,
                                 unsigned int* __restrict__ sp) {
  int row0 = blockIdx.x * 16;       // query-patch rows
  int col0 = blockIdx.y * 64;       // padded normal-patch cols
  v8f acc[4] = {};
  for (int k0 = 0; k0 < DD; k0 += 32) {
    v16h a = load_a_frag(A, DD, row0, k0);
#pragma unroll
    for (int j = 0; j < 4; ++j) {
      v16h b = load_b_frag(Bt, DD, col0 + j * 16, k0);
      acc[j] = __builtin_amdgcn_wmma_f32_16x16x32_f16(false, a, false, b,
                                                      (short)0, acc[j], false, false);
    }
  }
  int lane = threadIdx.x & 31;
  int cl = lane & 15;
  int rbase = (lane & 16) ? 8 : 0;
#pragma unroll
  for (int j = 0; j < 4; ++j) {
    int colg  = col0 + j * 16 + cl;
    int group = (col0 + j * 16) / PPAD;  // 16-wide tile never crosses an image (208 % 16 == 0)
    float sb = sqb[colg];
#pragma unroll
    for (int r = 0; r < 8; ++r) {
      int rowg = row0 + rbase + r;
      float d2 = sqa[rowg] + sb - 2.0f * acc[j][r];
      float d = sqrtf(fmaxf(d2, 1e-12f)) * 0.5f;
      // min across the 16 columns held by this half-wave (xor masks stay in-half)
#pragma unroll
      for (int m = 1; m < 16; m <<= 1)
        d = fminf(d, __shfl_xor(d, m, 32));
      if (cl == 0)  // d >= 0 -> float bits are order-preserving as uint
        atomicMin(&sp[(size_t)rowg * NN + group], __float_as_uint(d));
    }
  }
}

// ---------------- reductions + resize ----------------
__global__ void patch_map_kernel(const unsigned int* __restrict__ sp,
                                 float* __restrict__ pm) {
  int row = blockIdx.x * blockDim.x + threadIdx.x;
  if (row >= RA) return;
  float s = 0.f;
  for (int g = 0; g < NN; ++g) s += __uint_as_float(sp[(size_t)row * NN + g]);
  pm[row] = s * (1.0f / NN);
}

__global__ void scores_kernel(const unsigned int* __restrict__ sp,
                              float* __restrict__ out) {
  int n1 = blockIdx.x;
  int t = threadIdx.x;  // 64 = one thread per normal image
  __shared__ float red[64];
  float m = -1e30f;
  for (int p = 0; p < PPI; ++p)
    m = fmaxf(m, __uint_as_float(sp[((size_t)n1 * PPI + p) * NN + t]));
  red[t] = m; __syncthreads();
  for (int off = 32; off > 0; off >>= 1) {
    if (t < off) red[t] += red[t + off];
    __syncthreads();
  }
  if (t == 0) out[n1] = red[0] * (1.0f / NN);
}

__global__ void resize_kernel(const float* __restrict__ pm, float* __restrict__ out) {
  int idx = blockIdx.x * blockDim.x + threadIdx.x;
  if (idx >= NQ * HWH * HWH) return;
  int n = idx / (HWH * HWH);
  int rem = idx % (HWH * HWH);
  int y = rem / HWH, x = rem % HWH;
  const float scale = (float)PHD / (float)HWH;  // 1/16, half-pixel centers
  float sy = fmaxf((y + 0.5f) * scale - 0.5f, 0.0f);
  float sx = fmaxf((x + 0.5f) * scale - 0.5f, 0.0f);
  int y0 = (int)sy; int x0 = (int)sx;
  y0 = y0 < PHD - 1 ? y0 : PHD - 1;
  x0 = x0 < PHD - 1 ? x0 : PHD - 1;
  int y1 = y0 + 1 < PHD - 1 ? y0 + 1 : PHD - 1;
  int x1 = x0 + 1 < PHD - 1 ? x0 + 1 : PHD - 1;
  float fy = sy - (float)y0, fx = sx - (float)x0;
  const float* m = pm + (size_t)n * PPI;
  float v00 = m[y0 * PHD + x0], v01 = m[y0 * PHD + x1];
  float v10 = m[y1 * PHD + x0], v11 = m[y1 * PHD + x1];
  float v = v00 * (1.f - fy) * (1.f - fx) + v01 * (1.f - fy) * fx
          + v10 * fy * (1.f - fx)         + v11 * fy * fx;
  out[NQ + idx] = v;
}

// ---------------- host ----------------
extern "C" void kernel_launch(void* const* d_in, const int* in_sizes, int n_in,
                              void* d_out, int out_size, void* d_ws, size_t ws_size,
                              hipStream_t stream) {
  const float* images  = (const float*)d_in[0];
  const float* normals = (const float*)d_in[1];
  const float* W       = (const float*)d_in[2];
  float* out = (float*)d_out;

  char* ws = (char*)d_ws;
  size_t off = 0;
  auto alloc = [&](size_t bytes) -> void* {
    void* p = ws + off;
    off = (off + bytes + 255) & ~(size_t)255;
    return p;
  };
  _Float16*     Pall = (_Float16*)    alloc((size_t)RALL * DD * 2);
  _Float16*     Wt   = (_Float16*)    alloc((size_t)DD * DD * 2);
  float*        F    = (float*)       alloc((size_t)RALL * DD * 4);
  _Float16*     Af   = (_Float16*)    alloc((size_t)RA * DD * 2);
  _Float16*     Bpad = (_Float16*)    alloc((size_t)RBPAD * DD * 2);
  float*        sqa  = (float*)       alloc((size_t)RA * 4);
  float*        sqb  = (float*)       alloc((size_t)RBPAD * 4);
  unsigned int* sp   = (unsigned int*)alloc((size_t)RA * NN * 4);
  float*        pm   = (float*)       alloc((size_t)RA * 4);
  (void)ws_size; (void)in_sizes; (void)n_in; (void)out_size;

  init_kernel<<<2048, 256, 0, stream>>>(sp, sqb, Bpad);

  {
    size_t total = (size_t)RALL * DD;
    patchify_kernel<<<(unsigned)((total + 255) / 256), 256, 0, stream>>>(images, normals, Pall);
  }
  wtrans_kernel<<<(DD * DD + 255) / 256, 256, 0, stream>>>(W, Wt);

  proj_gemm_kernel<<<dim3(RALL / 16, DD / 64), 32, 0, stream>>>(Pall, Wt, F);

  normalize_kernel<<<RALL, 256, 0, stream>>>(F, Af, Bpad, sqa, sqb);

  dist_gemm_kernel<<<dim3(RA / 16, RBPAD / 64), 32, 0, stream>>>(Af, Bpad, sqa, sqb, sp);

  patch_map_kernel<<<(RA + 255) / 256, 256, 0, stream>>>(sp, pm);
  scores_kernel<<<NQ, 64, 0, stream>>>(sp, out);
  resize_kernel<<<(NQ * HWH * HWH + 255) / 256, 256, 0, stream>>>(pm, out);
}
